// EvolveGATO_25786983645429
// MI455X (gfx1250) — matrix operand order repaired and checked by reading, not compiled, not run on hardware
//
#include <hip/hip_runtime.h>
#include <hip/hip_bf16.h>
#include <stdint.h>

typedef __attribute__((ext_vector_type(16))) _Float16 v16h;
typedef __attribute__((ext_vector_type(8)))  float    v8f;

// ---------------------------------------------------------------------------
// CDNA5 async global->LDS copy (16B per lane) + wait.  ASYNCcnt tracked.
// ---------------------------------------------------------------------------
__device__ __forceinline__ void async_g2lds_b128(void* lds_dst, const void* gsrc) {
  uint32_t loff = (uint32_t)(uintptr_t)lds_dst;   // low 32 bits of LDS aperture addr == LDS byte offset
  asm volatile("global_load_async_to_lds_b128 %0, %1, off" :: "v"(loff), "v"(gsrc) : "memory");
}
__device__ __forceinline__ void async_wait0() {
  asm volatile("s_wait_asynccnt 0x0" ::: "memory");
}
__device__ __forceinline__ float sigm(float x) { return 1.0f / (1.0f + __expf(-x)); }

// ---------------------------------------------------------------------------
// 1) Evolve attention vector a via GRUCell(h=0) for 3 steps (both layers).
// ---------------------------------------------------------------------------
__global__ void evolve_a_k(const float* a0, const float* wih0, const float* bih0, const float* bhh0, float* out0,
                           const float* a1i, const float* wih1, const float* bih1, const float* bhh1, float* out1) {
  const float* ainit = blockIdx.x ? a1i : a0;
  const float* wih   = blockIdx.x ? wih1 : wih0;
  const float* bih   = blockIdx.x ? bih1 : bih0;
  const float* bhh   = blockIdx.x ? bhh1 : bhh0;
  float* aout        = blockIdx.x ? out1 : out0;
  __shared__ float ac[512];
  __shared__ float gi[1536];
  int tid = threadIdx.x;
  for (int k = tid; k < 512; k += 256) ac[k] = ainit[k];
  __syncthreads();
  for (int step = 0; step < 3; ++step) {
    for (int k = tid; k < 1536; k += 256) {
      float acc = bih[k];
      const float* wr = wih + (size_t)k * 512;
      for (int c = 0; c < 512; ++c) acc += ac[c] * wr[c];
      gi[k] = acc;
    }
    __syncthreads();
    for (int k = tid; k < 512; k += 256) {
      float r = sigm(gi[k] + bhh[k]);
      float z = sigm(gi[512 + k] + bhh[512 + k]);
      float n = tanhf(gi[1024 + k] + r * bhh[1024 + k]);
      ac[k] = (1.0f - z) * n;
    }
    __syncthreads();
  }
  for (int k = tid; k < 512; k += 256) aout[k] = ac[k];
}

// ---------------------------------------------------------------------------
// 2) Matrix-GRU step (gates then final).
// ---------------------------------------------------------------------------
__global__ void matgru_gates_k(const float* Q, const float* mW, const float* mU, const float* mb,
                               float* upd, float* rst, int R, int C) {
  int j = blockIdx.x * 64 + threadIdx.x;
  int i = blockIdx.y * 4 + threadIdx.y;
  if (i >= R || j >= C) return;
  float a0 = mb[(size_t)i * C + j];
  float a1 = mb[(size_t)R * C + (size_t)i * C + j];
  const float* w0 = mW + (size_t)i * R;
  const float* u0 = mU + (size_t)i * R;
  const float* w1 = mW + (size_t)R * R + (size_t)i * R;
  const float* u1 = mU + (size_t)R * R + (size_t)i * R;
  for (int k = 0; k < R; ++k) {
    float q = Q[(size_t)k * C + j];
    a0 += (w0[k] + u0[k]) * q;
    a1 += (w1[k] + u1[k]) * q;
  }
  upd[(size_t)i * C + j] = sigm(a0);
  rst[(size_t)i * C + j] = sigm(a1);
}

__global__ void matgru_final_k(const float* Q, const float* mW, const float* mU, const float* mb,
                               const float* upd, const float* rst, float* Qo, int R, int C) {
  int j = blockIdx.x * 64 + threadIdx.x;
  int i = blockIdx.y * 4 + threadIdx.y;
  if (i >= R || j >= C) return;
  float a = mb[(size_t)2 * R * C + (size_t)i * C + j];
  const float* w2 = mW + (size_t)2 * R * R + (size_t)i * R;
  const float* u2 = mU + (size_t)2 * R * R + (size_t)i * R;
  for (int k = 0; k < R; ++k) {
    float q = Q[(size_t)k * C + j];
    a += w2[k] * q + u2[k] * (rst[(size_t)k * C + j] * q);
  }
  float h = tanhf(a);
  float u = upd[(size_t)i * C + j];
  Qo[(size_t)i * C + j] = (1.0f - u) * Q[(size_t)i * C + j] + u * h;
}

// ---------------------------------------------------------------------------
// 3) f32 -> f16 with zero padding to (Rp, Cp)
// ---------------------------------------------------------------------------
__global__ void cvt_pad_k(const float* src, _Float16* dst, int R, int C, int Rp, int Cp) {
  int idx = blockIdx.x * blockDim.x + threadIdx.x;
  if (idx >= Rp * Cp) return;
  int r = idx / Cp, c = idx - r * Cp;
  float v = (r < R && c < C) ? src[(size_t)r * C + c] : 0.0f;
  dst[idx] = (_Float16)v;
}

// ---------------------------------------------------------------------------
// 4) Tiled WMMA GEMM: C(MxN,f32)[+optional f16 copy] = A(MxK,f16)*B(KxN,f16).
//    Double-buffered async global->LDS pipeline: wait(i) -> barrier ->
//    issue(i+1) -> compute(i), so copies overlap the WMMA work.
// ---------------------------------------------------------------------------
__global__ void gemm_wmma_k(const _Float16* __restrict__ A, const _Float16* __restrict__ B,
                            float* __restrict__ C, _Float16* __restrict__ Ch,
                            int M, int N, int K) {
  __shared__ __align__(64) _Float16 Al[2][128][32];
  __shared__ __align__(64) _Float16 Bl[2][32][64];
  int tid = threadIdx.x;
  int wave = tid >> 5, lane = tid & 31;
  int mbase = blockIdx.x * 128;
  int nbase = blockIdx.y * 64;
  int rowA = wave * 16 + (lane & 15);
  int half = lane >> 4;
  int ar0 = tid >> 2, ac0 = (tid & 3) * 8;
  int ar1 = (tid + 256) >> 2, ac1 = ((tid + 256) & 3) * 8;
  int br = tid >> 3, bc = (tid & 7) * 8;
  const _Float16* Ag0 = A + (size_t)(mbase + ar0) * K + ac0;
  const _Float16* Ag1 = A + (size_t)(mbase + ar1) * K + ac1;
  const _Float16* Bg  = B + (size_t)br * N + nbase + bc;
  async_g2lds_b128(&Al[0][ar0][ac0], Ag0);
  async_g2lds_b128(&Al[0][ar1][ac1], Ag1);
  async_g2lds_b128(&Bl[0][br][bc], Bg);
  v8f acc[4] = {};
  int nchunks = K >> 5;
  for (int i = 0; i < nchunks; ++i) {
    int b = i & 1;
    async_wait0();
    __syncthreads();
    if (i + 1 < nchunks) {
      int kb = (i + 1) * 32;
      async_g2lds_b128(&Al[b ^ 1][ar0][ac0], Ag0 + kb);
      async_g2lds_b128(&Al[b ^ 1][ar1][ac1], Ag1 + kb);
      async_g2lds_b128(&Bl[b ^ 1][br][bc], Bg + (size_t)kb * N);
    }
    v16h af = *(const v16h*)&Al[b][rowA][half * 16];
#pragma unroll
    for (int nt = 0; nt < 4; ++nt) {
      v16h bf;
#pragma unroll
      for (int s = 0; s < 16; ++s) bf[s] = Bl[b][half * 16 + s][nt * 16 + (lane & 15)];
      acc[nt] = __builtin_amdgcn_wmma_f32_16x16x32_f16(false, af, false, bf, (short)0, acc[nt], false, false);
    }
  }
  int rowo = mbase + wave * 16 + 8 * half;
  int colo = nbase + (lane & 15);
#pragma unroll
  for (int nt = 0; nt < 4; ++nt)
#pragma unroll
    for (int v = 0; v < 8; ++v) {
      float o = acc[nt][v];
      size_t idx = (size_t)(rowo + v) * N + colo + nt * 16;
      C[idx] = o;
      if (Ch) Ch[idx] = (_Float16)o;
    }
}

// ---------------------------------------------------------------------------
// 5) s_i = Wh_i . a[:256], t_i = Wh_i . a[256:]
// ---------------------------------------------------------------------------
__global__ void sv_k(const float* __restrict__ Wh, const float* __restrict__ av, float* s, float* tv) {
  int r = blockIdx.x * 256 + threadIdx.x;
  const float* row = Wh + (size_t)r * 256;
  float a = 0.f, b = 0.f;
  for (int c = 0; c < 256; ++c) { float w = row[c]; a += w * av[c]; b += w * av[256 + c]; }
  s[r] = a; tv[r] = b;
}

// ---------------------------------------------------------------------------
// 6) Per-row masked softmax stats (one wave per row).
// ---------------------------------------------------------------------------
__global__ void row_stats_k(const int* __restrict__ adj, const float* __restrict__ s,
                            const float* __restrict__ tv, float* m, float* invd) {
  int wave = threadIdx.x >> 5, lane = threadIdx.x & 31;
  int row = blockIdx.x * 8 + wave;
  const int* ar = adj + (size_t)row * 4096;
  float sr = s[row];
  float mx = -3.0e38f;
  for (int j = lane; j < 4096; j += 32) {
    __builtin_prefetch(ar + j + 512, 0, 0);
    if (ar[j] != 0 || j == row) {
      float v = sr + tv[j];
      v = v > 0.f ? v : 0.2f * v;
      mx = fmaxf(mx, v);
    }
  }
#pragma unroll
  for (int o = 16; o > 0; o >>= 1) mx = fmaxf(mx, __shfl_xor(mx, o, 32));
  float sum = 0.f;
  for (int j = lane; j < 4096; j += 32) {
    if (ar[j] != 0 || j == row) {
      float v = sr + tv[j];
      v = v > 0.f ? v : 0.2f * v;
      sum += __expf(v - mx);
    }
  }
#pragma unroll
  for (int o = 16; o > 0; o >>= 1) sum += __shfl_xor(sum, o, 32);
  if (lane == 0) { m[row] = mx; invd[row] = 1.0f / sum; }
}

// ---------------------------------------------------------------------------
// 7) Attention aggregation with double-buffered async Wh staging.
// ---------------------------------------------------------------------------
__global__ void att_agg_k(const int* __restrict__ adj, const float* __restrict__ s,
                          const float* __restrict__ tv, const float* __restrict__ m,
                          const float* __restrict__ invd, const _Float16* __restrict__ Whf,
                          _Float16* __restrict__ outh, int apply_elu) {
  __shared__ __align__(64) _Float16 WhLds[2][32][256];
  __shared__ float tl[2][32];
  int tid = threadIdx.x;
  int wave = tid >> 5, lane = tid & 31;
  int rbase = blockIdx.x * 128 + wave * 16;
  int half = lane >> 4;
  int r = rbase + (lane & 15);
  float s_r = s[r];
  float m_r = m[r];
  v8f acc[16] = {};
#pragma unroll
  for (int q = 0; q < 4; ++q) {
    int slot = tid + q * 256;
    int wr = slot >> 5, wc = (slot & 31) * 8;
    async_g2lds_b128(&WhLds[0][wr][wc], Whf + (size_t)wr * 256 + wc);
  }
  if (tid < 32) tl[0][tid] = tv[tid];
  for (int i = 0; i < 128; ++i) {
    int b = i & 1;
    int kb = i * 32;
    async_wait0();
    __syncthreads();
    if (i + 1 < 128) {
      int kn = kb + 32;
#pragma unroll
      for (int q = 0; q < 4; ++q) {
        int slot = tid + q * 256;
        int wr = slot >> 5, wc = (slot & 31) * 8;
        async_g2lds_b128(&WhLds[b ^ 1][wr][wc], Whf + (size_t)(kn + wr) * 256 + wc);
      }
      if (tid < 32) tl[b ^ 1][tid] = tv[kn + tid];
    }
    int am[16];
    const int4* ap = (const int4*)(adj + (size_t)r * 4096 + kb + half * 16);
#pragma unroll
    for (int q = 0; q < 4; ++q) ((int4*)am)[q] = ap[q];
    v16h af;
#pragma unroll
    for (int e = 0; e < 16; ++e) {
      int j = kb + half * 16 + e;
      float v = s_r + tl[b][half * 16 + e];
      v = v > 0.f ? v : 0.2f * v;
      float p = (am[e] != 0 || j == r) ? __expf(v - m_r) : 0.0f;
      af[e] = (_Float16)p;
    }
#pragma unroll
    for (int nt = 0; nt < 16; ++nt) {
      v16h bf;
#pragma unroll
      for (int s = 0; s < 16; ++s) bf[s] = WhLds[b][half * 16 + s][nt * 16 + (lane & 15)];
      acc[nt] = __builtin_amdgcn_wmma_f32_16x16x32_f16(false, af, false, bf, (short)0, acc[nt], false, false);
    }
  }
  float sc[8];
  int row0 = rbase + 8 * half;
#pragma unroll
  for (int v = 0; v < 8; ++v) sc[v] = invd[row0 + v];
  int col0 = lane & 15;
#pragma unroll
  for (int nt = 0; nt < 16; ++nt)
#pragma unroll
    for (int v = 0; v < 8; ++v) {
      float o = acc[nt][v] * sc[v];
      if (apply_elu && o < 0.f) o = __expf(o) - 1.0f;
      outh[(size_t)(row0 + v) * 256 + nt * 16 + col0] = (_Float16)o;
    }
}

// ---------------------------------------------------------------------------
// 8) Classifier tail
// ---------------------------------------------------------------------------
__global__ void cls_final_k(const float* __restrict__ hid, const float* __restrict__ b1,
                            const float* __restrict__ w2, const float* __restrict__ b2,
                            float* __restrict__ out) {
  int r = blockIdx.x * 256 + threadIdx.x;
  float a0 = b2[0], a1 = b2[1];
  const float* hr = hid + (size_t)r * 320;
  for (int h = 0; h < 307; ++h) {
    float v = hr[h] + b1[h];
    v = v > 0.f ? v : 0.f;
    a0 += v * w2[2 * h];
    a1 += v * w2[2 * h + 1];
  }
  out[2 * r] = a0;
  out[2 * r + 1] = a1;
}

// ---------------------------------------------------------------------------
// Host orchestration
// ---------------------------------------------------------------------------
extern "C" void kernel_launch(void* const* d_in, const int* in_sizes, int n_in,
                              void* d_out, int out_size, void* d_ws, size_t ws_size,
                              hipStream_t stream) {
  (void)in_sizes; (void)n_in; (void)out_size; (void)ws_size;
  const float* feats = (const float*)d_in[0];
  const int*   adj   = (const int*)d_in[1];
  const float* W0    = (const float*)d_in[2];
  const float* a0    = (const float*)d_in[3];
  const float* mg0W  = (const float*)d_in[4];
  const float* mg0U  = (const float*)d_in[5];
  const float* mg0b  = (const float*)d_in[6];
  const float* g0wih = (const float*)d_in[7];
  const float* g0bih = (const float*)d_in[9];
  const float* g0bhh = (const float*)d_in[10];
  const float* W1    = (const float*)d_in[11];
  const float* a1    = (const float*)d_in[12];
  const float* mg1W  = (const float*)d_in[13];
  const float* mg1U  = (const float*)d_in[14];
  const float* mg1b  = (const float*)d_in[15];
  const float* g1wih = (const float*)d_in[16];
  const float* g1bih = (const float*)d_in[18];
  const float* g1bhh = (const float*)d_in[19];
  const float* w1    = (const float*)d_in[20];
  const float* b1    = (const float*)d_in[21];
  const float* w2    = (const float*)d_in[22];
  const float* b2    = (const float*)d_in[23];

  const float* feats2 = feats + (size_t)2 * 4096 * 166;  // only t=2 is observable
  const int*   adj2   = adj + (size_t)2 * 4096 * 4096;

  char* ws = (char*)d_ws;
  size_t off = 0;
  auto alloc = [&](size_t bytes) -> void* {
    void* p = ws + off;
    off += (bytes + 255) & ~(size_t)255;
    return p;
  };
  float* aF0 = (float*)alloc(512 * 4);
  float* aF1 = (float*)alloc(512 * 4);
  float* Q0A = (float*)alloc((size_t)166 * 256 * 4);
  float* Q0B = (float*)alloc((size_t)166 * 256 * 4);
  float* U0  = (float*)alloc((size_t)166 * 256 * 4);
  float* R0  = (float*)alloc((size_t)166 * 256 * 4);
  float* Q1A = (float*)alloc((size_t)256 * 256 * 4);
  float* Q1B = (float*)alloc((size_t)256 * 256 * 4);
  float* U1  = (float*)alloc((size_t)256 * 256 * 4);
  float* R1  = (float*)alloc((size_t)256 * 256 * 4);
  _Float16* Xh  = (_Float16*)alloc((size_t)4096 * 192 * 2);
  _Float16* W0h = (_Float16*)alloc((size_t)192 * 256 * 2);
  _Float16* W1h = (_Float16*)alloc((size_t)256 * 256 * 2);
  float*    Wh  = (float*)alloc((size_t)4096 * 256 * 4);
  _Float16* Whh = (_Float16*)alloc((size_t)4096 * 256 * 2);
  float* svec = (float*)alloc(4096 * 4);
  float* tvec = (float*)alloc(4096 * 4);
  float* mvec = (float*)alloc(4096 * 4);
  float* dvec = (float*)alloc(4096 * 4);
  _Float16* h0h = (_Float16*)alloc((size_t)4096 * 256 * 2);
  _Float16* h1h = (_Float16*)alloc((size_t)4096 * 256 * 2);
  _Float16* w1h = (_Float16*)alloc((size_t)256 * 320 * 2);
  float*    hid = (float*)alloc((size_t)4096 * 320 * 4);

  evolve_a_k<<<2, 256, 0, stream>>>(a0, g0wih, g0bih, g0bhh, aF0,
                                    a1, g1wih, g1bih, g1bhh, aF1);

  dim3 mblk(64, 4);
  dim3 g0((256 + 63) / 64, (166 + 3) / 4);
  matgru_gates_k<<<g0, mblk, 0, stream>>>(W0, mg0W, mg0U, mg0b, U0, R0, 166, 256);
  matgru_final_k<<<g0, mblk, 0, stream>>>(W0, mg0W, mg0U, mg0b, U0, R0, Q0A, 166, 256);
  matgru_gates_k<<<g0, mblk, 0, stream>>>(Q0A, mg0W, mg0U, mg0b, U0, R0, 166, 256);
  matgru_final_k<<<g0, mblk, 0, stream>>>(Q0A, mg0W, mg0U, mg0b, U0, R0, Q0B, 166, 256);
  matgru_gates_k<<<g0, mblk, 0, stream>>>(Q0B, mg0W, mg0U, mg0b, U0, R0, 166, 256);
  matgru_final_k<<<g0, mblk, 0, stream>>>(Q0B, mg0W, mg0U, mg0b, U0, R0, Q0A, 166, 256);
  dim3 g1g((256 + 63) / 64, (256 + 3) / 4);
  matgru_gates_k<<<g1g, mblk, 0, stream>>>(W1, mg1W, mg1U, mg1b, U1, R1, 256, 256);
  matgru_final_k<<<g1g, mblk, 0, stream>>>(W1, mg1W, mg1U, mg1b, U1, R1, Q1A, 256, 256);
  matgru_gates_k<<<g1g, mblk, 0, stream>>>(Q1A, mg1W, mg1U, mg1b, U1, R1, 256, 256);
  matgru_final_k<<<g1g, mblk, 0, stream>>>(Q1A, mg1W, mg1U, mg1b, U1, R1, Q1B, 256, 256);
  matgru_gates_k<<<g1g, mblk, 0, stream>>>(Q1B, mg1W, mg1U, mg1b, U1, R1, 256, 256);
  matgru_final_k<<<g1g, mblk, 0, stream>>>(Q1B, mg1W, mg1U, mg1b, U1, R1, Q1A, 256, 256);

  // --- layer 0 GAT at t=2 ---
  cvt_pad_k<<<(4096 * 192 + 255) / 256, 256, 0, stream>>>(feats2, Xh, 4096, 166, 4096, 192);
  cvt_pad_k<<<(192 * 256 + 255) / 256, 256, 0, stream>>>(Q0A, W0h, 166, 256, 192, 256);
  gemm_wmma_k<<<dim3(32, 4), 256, 0, stream>>>(Xh, W0h, Wh, Whh, 4096, 256, 192);
  sv_k<<<16, 256, 0, stream>>>(Wh, aF0, svec, tvec);
  row_stats_k<<<512, 256, 0, stream>>>(adj2, svec, tvec, mvec, dvec);
  att_agg_k<<<32, 256, 0, stream>>>(adj2, svec, tvec, mvec, dvec, Whh, h0h, 0);

  // --- layer 1 GAT at t=2 (ELU epilogue) ---
  cvt_pad_k<<<(256 * 256 + 255) / 256, 256, 0, stream>>>(Q1A, W1h, 256, 256, 256, 256);
  gemm_wmma_k<<<dim3(32, 4), 256, 0, stream>>>(h0h, W1h, Wh, Whh, 4096, 256, 256);
  sv_k<<<16, 256, 0, stream>>>(Wh, aF1, svec, tvec);
  row_stats_k<<<512, 256, 0, stream>>>(adj2, svec, tvec, mvec, dvec);
  att_agg_k<<<32, 256, 0, stream>>>(adj2, svec, tvec, mvec, dvec, Whh, h1h, 1);

  // --- classifier ---
  cvt_pad_k<<<(256 * 320 + 255) / 256, 256, 0, stream>>>(w1, w1h, 256, 307, 256, 320);
  gemm_wmma_k<<<dim3(32, 5), 256, 0, stream>>>(h1h, w1h, hid, (_Float16*)nullptr, 4096, 320, 256);
  cls_final_k<<<16, 256, 0, stream>>>(hid, b1, w2, b2, (float*)d_out);
}
